// ContactPerceiverWithMamba_962072675000
// MI455X (gfx1250) — compile-verified, matrix operand order checked by reading, not run on hardware
//
#include <hip/hip_runtime.h>
#include <hip/hip_bf16.h>
#include <cstdint>

// ---------------------------------------------------------------------------
// ContactPerceiverWithMamba — CDNA5 (gfx1250) implementation.
// Large GEMMs run through v_wmma_f32_16x16x32_f16 (wave32 WMMA).
// Weights pre-converted to dense f16 once per call; weight tiles stream to
// LDS via GLOBAL_LOAD_ASYNC_TO_LDS_B128 (ASYNCcnt), activations convert
// f32->f16 while staging. Accumulation in f32. Block tile 128x128, wave
// tile 32x64 (8 WMMAs per K-step per wave).
// ---------------------------------------------------------------------------

typedef __attribute__((ext_vector_type(16))) _Float16 v16h;
typedef __attribute__((ext_vector_type(8)))  float    v8f;

namespace cpm {
constexpr int Bb     = 4;
constexpr int Nn     = 8192;
constexpr int LTXT   = 77;
constexpr int Cc     = 512;
constexpr int HEADS  = 8;
constexpr int LQ     = LTXT + 1;       // 78
constexpr int MBIG   = Bb * Nn;        // 32768
constexpr int MSMALL = Bb * LQ;        // 312
constexpr int DINNER = 1024;
constexpr int DSTATE = 16;
constexpr int DTRANK = 32;
}

// ------------------------------ device helpers ------------------------------

__device__ __forceinline__ float gelu_tanh(float x) {
  const float c0 = 0.7978845608028654f, c1 = 0.044715f;
  return 0.5f * x * (1.0f + tanhf(c0 * (x + c1 * x * x * x)));
}
__device__ __forceinline__ float silu_f(float x) {
  return x / (1.0f + __expf(-x));
}
__device__ __forceinline__ float softplus_f(float x) {
  return (x > 20.0f) ? x : log1pf(__expf(x));
}

// async 16-byte global -> LDS copy (per-lane), tracked by ASYNCcnt
__device__ __forceinline__ void async_g2l_b128(uint32_t lds_off,
                                               const void* gaddr) {
  asm volatile("global_load_async_to_lds_b128 %0, %1, off"
               :: "v"(lds_off), "v"(gaddr)
               : "memory");
}
__device__ __forceinline__ void wait_async0() {
  asm volatile("s_wait_asynccnt 0x0" ::: "memory");
}

// Load one 16x32 f16 fragment (A-layout, ISA 7.12.2) from an LDS tile with
// leading dimension 32. lane 0-15 -> row r; VGPR pair v holds K pairs; lanes
// 16-31 take the +8 K offset within each half.
__device__ __forceinline__ v16h frag_ld(const _Float16* __restrict__ s,
                                        int row0, int lane) {
  const _Float16* p = s + (size_t)(row0 + (lane & 15)) * 32;
  const int hsel = (lane >> 4) ? 8 : 0;
  v16h f;
#pragma unroll
  for (int v = 0; v < 8; ++v) {
    int kb = ((v & 4) ? 16 : 0) + hsel + ((v & 3) << 1);
    f[2 * v]     = p[kb];
    f[2 * v + 1] = p[kb + 1];
  }
  return f;
}

// -------------------------- weight f32 -> f16 repack ------------------------
// out[n*K + k] = (f16) W[n*ldw + wcol + k]   (dense, 16B-aligned rows)
__global__ __launch_bounds__(256)
void k_cvt_w(const float* __restrict__ W, int ldw, int wcol,
             _Float16* __restrict__ out, long total, int K) {
  long i = (long)blockIdx.x * 256 + threadIdx.x;
  long stride = (long)gridDim.x * 256;
  for (; i < total; i += stride) {
    long n = i / K, k = i - n * K;
    out[i] = (_Float16)W[n * (long)ldw + wcol + k];
  }
}

// ------------------------------ WMMA GEMM -----------------------------------
// out[M,N] = res? + act( A[:, acol:acol+K] @ W16[N,K]^T + bias )
// A f32 row-major lda; W16 dense f16 [N][K]. N % 64 == 0, K % 32 == 0.
// Block tile 128x128, 8 waves (4x2), wave tile 32x64.
// act: 0 none, 1 gelu, 2 silu, 3 softplus.
__global__ __launch_bounds__(256)
void k_gemm_wmma(const float* __restrict__ A, int lda, int acol,
                 const _Float16* __restrict__ W16,
                 const float* __restrict__ bias,
                 const float* __restrict__ res, int ldr,
                 float* __restrict__ out, int ldo,
                 int M, int N, int K, int act) {
  __shared__ _Float16 As[128 * 32];
  __shared__ _Float16 Bs[128 * 32];

  const int tid  = threadIdx.x;
  const int lane = tid & 31;
  const int wid  = tid >> 5;    // 0..7
  const int wm   = wid >> 1;    // 0..3  (M direction, 32 rows each)
  const int wn   = wid & 1;     // 0..1  (N direction, 64 cols each)
  const int m_blk = blockIdx.y * 128;
  const int n_blk = blockIdx.x * 128;

  v8f acc[8];
#pragma unroll
  for (int i = 0; i < 8; ++i)
#pragma unroll
    for (int j = 0; j < 8; ++j) acc[i][j] = 0.0f;

  for (int k0 = 0; k0 < K; k0 += 32) {
    // stage W tile (128x32 f16) via async global->LDS; 2 segments/thread
#pragma unroll
    for (int s = 0; s < 2; ++s) {
      int seg = tid + s * 256;       // 0..511
      int r  = seg >> 2;             // 0..127
      int cb = (seg & 3) * 8;        // halves (16B)
      int gn = n_blk + r;
      if (gn < N) {
        uint32_t loff = (uint32_t)(size_t)(Bs + r * 32 + cb);
        async_g2l_b128(loff, (const void*)(W16 + (size_t)gn * K + k0 + cb));
      }
    }
    // stage A tile (128x32): each thread converts 16 contiguous f32 -> f16
    {
      int r = tid >> 1;
      int cb = (tid & 1) * 16;
      int gm = m_blk + r;
      _Float16* dst = As + r * 32 + cb;
      if (gm < M) {
        const float* src = A + (size_t)gm * lda + acol + k0 + cb;
        if (k0 + 32 < K) __builtin_prefetch(src + 32, 0, 1);
#pragma unroll
        for (int c = 0; c < 16; ++c) dst[c] = (_Float16)src[c];
      } else {
#pragma unroll
        for (int c = 0; c < 16; ++c) dst[c] = (_Float16)0.0f;
      }
    }
    wait_async0();
    __syncthreads();

    v16h a0 = frag_ld(As, wm * 32,      lane);
    v16h a1 = frag_ld(As, wm * 32 + 16, lane);
    v16h b0 = frag_ld(Bs, wn * 64,      lane);
    v16h b1 = frag_ld(Bs, wn * 64 + 16, lane);
    v16h b2 = frag_ld(Bs, wn * 64 + 32, lane);
    v16h b3 = frag_ld(Bs, wn * 64 + 48, lane);

    acc[0] = __builtin_amdgcn_wmma_f32_16x16x32_f16(false, a0, false, b0,
                                                    (short)0, acc[0], false, false);
    acc[1] = __builtin_amdgcn_wmma_f32_16x16x32_f16(false, a0, false, b1,
                                                    (short)0, acc[1], false, false);
    acc[2] = __builtin_amdgcn_wmma_f32_16x16x32_f16(false, a0, false, b2,
                                                    (short)0, acc[2], false, false);
    acc[3] = __builtin_amdgcn_wmma_f32_16x16x32_f16(false, a0, false, b3,
                                                    (short)0, acc[3], false, false);
    acc[4] = __builtin_amdgcn_wmma_f32_16x16x32_f16(false, a1, false, b0,
                                                    (short)0, acc[4], false, false);
    acc[5] = __builtin_amdgcn_wmma_f32_16x16x32_f16(false, a1, false, b1,
                                                    (short)0, acc[5], false, false);
    acc[6] = __builtin_amdgcn_wmma_f32_16x16x32_f16(false, a1, false, b2,
                                                    (short)0, acc[6], false, false);
    acc[7] = __builtin_amdgcn_wmma_f32_16x16x32_f16(false, a1, false, b3,
                                                    (short)0, acc[7], false, false);
    __syncthreads();
  }

  // epilogue: C/D layout — VGPR v: lanes 0-15 row v, lanes 16-31 row 8+v;
  // column = lane & 15.  acc[t]: t>>2 selects A frag (16 M), t&3 B frag (16 N)
#pragma unroll
  for (int t = 0; t < 8; ++t) {
    int mb = m_blk + wm * 32 + (t >> 2) * 16;
    int nb = n_blk + wn * 64 + (t & 3) * 16;
    int nn = nb + (lane & 15);
    int mo = (lane >> 4) * 8;
    if (nn < N) {
      float bv = (bias != nullptr) ? bias[nn] : 0.0f;
#pragma unroll
      for (int v = 0; v < 8; ++v) {
        int mm = mb + mo + v;
        if (mm < M) {
          float val = acc[t][v] + bv;
          if (act == 1)      val = gelu_tanh(val);
          else if (act == 2) val = silu_f(val);
          else if (act == 3) val = softplus_f(val);
          if (res != nullptr) val += res[(size_t)mm * ldr + nn];
          out[(size_t)mm * ldo + nn] = val;
        }
      }
    }
  }
}

// --------------------------- encoder rank-4 fixup ---------------------------
// enc_kv[row, c] += x[row]*W[c,0] + xyz[row,:] . W[c,513:516]
__global__ __launch_bounds__(256)
void k_encode_fixup(const float* __restrict__ x, const float* __restrict__ xyz,
                    const float* __restrict__ Wt, float* __restrict__ out) {
  int row = blockIdx.x;
  float xv = x[row];
  float p0 = xyz[row * 3 + 0], p1 = xyz[row * 3 + 1], p2 = xyz[row * 3 + 2];
#pragma unroll
  for (int rep = 0; rep < 2; ++rep) {
    int c = threadIdx.x + rep * 256;
    const float* wr = Wt + (size_t)c * 516;
    out[(size_t)row * 512 + c] +=
        xv * wr[0] + p0 * wr[513] + p1 * wr[514] + p2 * wr[515];
  }
}

// -------------------------------- LayerNorm ---------------------------------
// one wave32 per 512-wide row, 8 rows per block
__global__ __launch_bounds__(256)
void k_layernorm(const float* __restrict__ in, const float* __restrict__ g,
                 const float* __restrict__ b, float* __restrict__ out, int rows) {
  int wid = threadIdx.x >> 5, lane = threadIdx.x & 31;
  int row = blockIdx.x * 8 + wid;
  if (row >= rows) return;
  const float* p = in + (size_t)row * 512;
  float v[16];
  float s = 0.0f;
#pragma unroll
  for (int i = 0; i < 16; ++i) { v[i] = p[lane * 16 + i]; s += v[i]; }
#pragma unroll
  for (int off = 16; off; off >>= 1) s += __shfl_xor(s, off, 32);
  float mean = s * (1.0f / 512.0f);
  float vs = 0.0f;
#pragma unroll
  for (int i = 0; i < 16; ++i) { float d = v[i] - mean; vs += d * d; }
#pragma unroll
  for (int off = 16; off; off >>= 1) vs += __shfl_xor(vs, off, 32);
  float rstd = rsqrtf(vs * (1.0f / 512.0f) + 1e-5f);
  float* q = out + (size_t)row * 512;
#pragma unroll
  for (int i = 0; i < 16; ++i) {
    int c = lane * 16 + i;
    q[c] = (v[i] - mean) * rstd * g[c] + b[c];
  }
}

// ---------------------- attention, few queries x 8192 keys ------------------
// one block per (b, h, qi); scores buffered in LDS (8192 * 4B = 32KB)
__global__ __launch_bounds__(256)
void k_attn_qsmall(const float* __restrict__ q, const float* __restrict__ k,
                   const float* __restrict__ v, float* __restrict__ o,
                   int Lq, int Lk) {
  __shared__ float sc[8192];
  __shared__ float red[256];
  __shared__ float oacc[4][64];
  __shared__ float qrow[64];
  int idx = blockIdx.x;
  int qi = idx % Lq;
  int bh = idx / Lq;
  int h = bh % cpm::HEADS;
  int b = bh / cpm::HEADS;
  int tid = threadIdx.x;

  if (tid < 64)
    qrow[tid] = q[((size_t)(b * Lq + qi)) * 512 + h * 64 + tid] * 0.125f;
  __syncthreads();

  float lmax = -1e30f;
  for (int key = tid; key < Lk; key += 256) {
    const float* kp = k + ((size_t)(b * Lk + key)) * 512 + h * 64;
    float s = 0.0f;
#pragma unroll
    for (int d = 0; d < 64; ++d) s += qrow[d] * kp[d];
    sc[key] = s;
    lmax = fmaxf(lmax, s);
  }
  red[tid] = lmax;
  __syncthreads();
  for (int st = 128; st; st >>= 1) {
    if (tid < st) red[tid] = fmaxf(red[tid], red[tid + st]);
    __syncthreads();
  }
  float m = red[0];
  __syncthreads();

  float lsum = 0.0f;
  for (int key = tid; key < Lk; key += 256) {
    float e = __expf(sc[key] - m);
    sc[key] = e;
    lsum += e;
  }
  red[tid] = lsum;
  __syncthreads();
  for (int st = 128; st; st >>= 1) {
    if (tid < st) red[tid] += red[tid + st];
    __syncthreads();
  }
  float inv = 1.0f / red[0];
  __syncthreads();

  int d = tid & 63, gsel = tid >> 6;
  float a = 0.0f;
  for (int key = gsel; key < Lk; key += 4)
    a += sc[key] * v[((size_t)(b * Lk + key)) * 512 + h * 64 + d];
  oacc[gsel][d] = a;
  __syncthreads();
  if (tid < 64) {
    float s = (oacc[0][tid] + oacc[1][tid] + oacc[2][tid] + oacc[3][tid]) * inv;
    o[((size_t)(b * Lq + qi)) * 512 + h * 64 + tid] = s;
  }
}

// ---------------------- attention, 8192 queries x 78 keys -------------------
// block = (b, h, chunk of 128 q rows); K/V tiles in LDS; per-thread 2-pass
// softmax (max pass, then exp/sum/AV pass).
__global__ __launch_bounds__(128)
void k_attn_ksmall(const float* __restrict__ q, const float* __restrict__ k,
                   const float* __restrict__ v, float* __restrict__ o,
                   int Lq, int Lk) {
  __shared__ float sk[cpm::LQ * 64];
  __shared__ float sv[cpm::LQ * 64];
  int blocksPerBH = Lq >> 7;
  int idx = blockIdx.x;
  int chunk = idx % blocksPerBH;
  int bh = idx / blocksPerBH;
  int h = bh % cpm::HEADS;
  int b = bh / cpm::HEADS;
  int tid = threadIdx.x;

  for (int i = tid; i < Lk * 64; i += 128) {
    int key = i >> 6, d = i & 63;
    sk[i] = k[((size_t)(b * Lk + key)) * 512 + h * 64 + d];
    sv[i] = v[((size_t)(b * Lk + key)) * 512 + h * 64 + d];
  }
  __syncthreads();

  int qi = chunk * 128 + tid;
  const float* qp = q + ((size_t)(b * Lq + qi)) * 512 + h * 64;
  float qr[64];
#pragma unroll
  for (int d = 0; d < 64; ++d) qr[d] = qp[d] * 0.125f;

  float m = -1e30f;
  for (int key = 0; key < Lk; ++key) {
    const float* kp = sk + key * 64;
    float s = 0.0f;
#pragma unroll
    for (int d = 0; d < 64; ++d) s += qr[d] * kp[d];
    m = fmaxf(m, s);
  }
  float oa[64];
#pragma unroll
  for (int d = 0; d < 64; ++d) oa[d] = 0.0f;
  float sum = 0.0f;
  for (int key = 0; key < Lk; ++key) {
    const float* kp = sk + key * 64;
    float s = 0.0f;
#pragma unroll
    for (int d = 0; d < 64; ++d) s += qr[d] * kp[d];
    float e = __expf(s - m);
    sum += e;
    const float* vp = sv + key * 64;
#pragma unroll
    for (int d = 0; d < 64; ++d) oa[d] += e * vp[d];
  }
  float inv = 1.0f / sum;
  float* op = o + ((size_t)(b * Lq + qi)) * 512 + h * 64;
#pragma unroll
  for (int d = 0; d < 64; ++d) op[d] = oa[d] * inv;
}

// --------------------------- mamba depthwise conv ---------------------------
__global__ __launch_bounds__(256)
void k_mamba_conv(const float* __restrict__ xz, const float* __restrict__ cw,
                  const float* __restrict__ cb, float* __restrict__ xc) {
  int b = blockIdx.x >> 2;
  int d = ((blockIdx.x & 3) << 8) + threadIdx.x;
  float w0 = cw[d * 4 + 0], w1 = cw[d * 4 + 1];
  float w2 = cw[d * 4 + 2], w3 = cw[d * 4 + 3];
  float bb = cb[d];
  float x0 = 0.0f, x1 = 0.0f, x2 = 0.0f, x3;
  for (int l = 0; l < cpm::LQ; ++l) {
    x3 = xz[((size_t)(b * cpm::LQ + l)) * 2048 + d];
    float s = x0 * w0 + x1 * w1 + x2 * w2 + x3 * w3 + bb;
    xc[((size_t)(b * cpm::LQ + l)) * 1024 + d] = silu_f(s);
    x0 = x1; x1 = x2; x2 = x3;
  }
}

// ---------------------------- mamba selective scan --------------------------
__global__ __launch_bounds__(256)
void k_mamba_scan(const float* __restrict__ xc, const float* __restrict__ xdbl,
                  const float* __restrict__ dt, const float* __restrict__ xz,
                  const float* __restrict__ Alog, const float* __restrict__ Dp,
                  float* __restrict__ y) {
  int b = blockIdx.x >> 2;
  int d = ((blockIdx.x & 3) << 8) + threadIdx.x;
  float a[cpm::DSTATE], h[cpm::DSTATE];
#pragma unroll
  for (int s = 0; s < cpm::DSTATE; ++s) {
    a[s] = -__expf(Alog[(size_t)d * cpm::DSTATE + s]);
    h[s] = 0.0f;
  }
  float Dv = Dp[d];
  for (int l = 0; l < cpm::LQ; ++l) {
    size_t row = (size_t)(b * cpm::LQ + l);
    float dtv = dt[row * 1024 + d];
    float xv  = xc[row * 1024 + d];
    float yv = 0.0f;
#pragma unroll
    for (int s = 0; s < cpm::DSTATE; ++s) {
      float Bv = xdbl[row * 64 + cpm::DTRANK + s];
      float Cv = xdbl[row * 64 + cpm::DTRANK + cpm::DSTATE + s];
      h[s] = __expf(dtv * a[s]) * h[s] + dtv * Bv * xv;
      yv += h[s] * Cv;
    }
    yv += Dv * xv;
    float zv = xz[row * 2048 + 1024 + d];
    y[row * 1024 + d] = yv * silu_f(zv);
  }
}

// ------------------------------ host sequencing -----------------------------

enum InIdx {
  IN_X = 0, IN_PF, IN_LANG, IN_TEMB, IN_XYZ,
  P_ENC_AD_W, P_ENC_AD_B, P_LANG_W, P_LANG_B, P_TIME_W, P_TIME_B,
  P_DEC_AD_W, P_DEC_AD_B,
  EA_QLNG, EA_QLNB, EA_KVLNG, EA_KVLNB, EA_WQ, EA_BQ, EA_WK, EA_BK,
  EA_WV, EA_BV, EA_WO, EA_BO, EA_MLNG, EA_MLNB, EA_W1, EA_B1, EA_W2, EA_B2,
  DA_QLNG, DA_QLNB, DA_KVLNG, DA_KVLNB, DA_WQ, DA_BQ, DA_WK, DA_BK,
  DA_WV, DA_BV, DA_WO, DA_BO, DA_MLNG, DA_MLNB, DA_W1, DA_B1, DA_W2, DA_B2,
  M_INW, M_CONVW, M_CONVB, M_XPROJW, M_DTW, M_DTB, M_ALOG, M_DD, M_OUTW
};

static inline void gemm(hipStream_t st, const float* A, int lda, int acol,
                        const _Float16* W16, const float* bias,
                        const float* res, int ldr,
                        float* out, int ldo, int M, int N, int K, int act) {
  dim3 g((N + 127) / 128, (M + 127) / 128);
  k_gemm_wmma<<<g, dim3(256), 0, st>>>(A, lda, acol, W16, bias, res, ldr,
                                       out, ldo, M, N, K, act);
}

static inline void lnorm(hipStream_t st, const float* in, const float* g,
                         const float* b, float* out, int rows) {
  k_layernorm<<<dim3((rows + 7) / 8), dim3(256), 0, st>>>(in, g, b, out, rows);
}

extern "C" void kernel_launch(void* const* d_in, const int* in_sizes, int n_in,
                              void* d_out, int out_size, void* d_ws, size_t ws_size,
                              hipStream_t stream) {
  (void)in_sizes; (void)n_in; (void)out_size; (void)ws_size;
  using namespace cpm;
#define FIN(i) ((const float*)d_in[(i)])

  float* dout = (float*)d_out;
  float* WS = (float*)d_ws;
  const size_t SLOT = (size_t)MBIG * Cc;           // 16,777,216 floats
  float* S0 = WS;                                  // enc_kv  -> h2
  float* S1 = WS + SLOT;                           // kvn     -> dec_q
  float* S2 = WS + 2 * SLOT;                       // k_enc   -> qn2 -> o2
  float* S3 = WS + 3 * SLOT;                       // v_enc   -> q2  -> mlp_ln
  float* S4 = WS + 4 * SLOT;                       // dec MLP hidden chunk
  float* SM = WS + 5 * SLOT;                       // small-tensor arena

  float* cur    = SM;                              // enc_q        [312,512]
  float* tmpa   = SM + 163840;                     // LN scratch   [312,512]
  float* tmpb   = SM + 327680;                     // q proj       [312,512]
  float* tmpc   = SM + 491520;                     // attn out     [312,512]
  float* tmph   = SM + 655360;                     // residual h   [312,512]
  float* tmpm   = SM + 819200;                     // mlp LN       [312,512]
  float* tmpw   = SM + 983040;                     // mlp hidden   [312,2048]
  float* kv2k   = SM + 1703936;                    // dec K        [312,512]
  float* kv2v   = SM + 1867776;                    // dec V        [312,512]
  float* xzbuf  = SM + 2031616;                    // mamba xz     [312,2048]
  float* xcbuf  = SM + 2703360;                    // mamba conv   [312,1024]
  float* xdblbf = SM + 3047424;                    // mamba xdbl   [312,64]
  float* dtbuf  = SM + 3145728;                    // mamba dt     [312,1024]
  float* ybuf   = SM + 3538944;                    // mamba y      [312,1024]
  float* curnew = SM + 3932160;                    // mamba pingpong [312,512]

  // f16 weight arena (~14.1M halves) after the small arena
  _Float16* wcur = (_Float16*)(SM + 4194304);
  auto cvt = [&](int idx, int ldw, int wcol, int N, int K) -> const _Float16* {
    _Float16* dst = wcur;
    long total = (long)N * K;
    int grid = (int)((total + 255) / 256);
    if (grid > 2048) grid = 2048;
    k_cvt_w<<<dim3(grid), dim3(256), 0, stream>>>(FIN(idx), ldw, wcol,
                                                  dst, total, K);
    wcur += total;
    return dst;
  };

  // ---- one-time (per call) f16 weight repacks ----
  const _Float16* w16_enc_ad = cvt(P_ENC_AD_W, 516, 1, 512, 512);
  const _Float16* w16_lang   = cvt(P_LANG_W,   512, 0, 512, 512);
  const _Float16* w16_time   = cvt(P_TIME_W,   512, 0, 512, 512);
  const _Float16* w16_dec_ad = cvt(P_DEC_AD_W, 512, 0, 512, 512);
  const _Float16* w16_ea_q   = cvt(EA_WQ, 512, 0, 512, 512);
  const _Float16* w16_ea_k   = cvt(EA_WK, 512, 0, 512, 512);
  const _Float16* w16_ea_v   = cvt(EA_WV, 512, 0, 512, 512);
  const _Float16* w16_ea_o   = cvt(EA_WO, 512, 0, 512, 512);
  const _Float16* w16_ea_1   = cvt(EA_W1, 512, 0, 2048, 512);
  const _Float16* w16_ea_2   = cvt(EA_W2, 2048, 0, 512, 2048);
  const _Float16* w16_da_q   = cvt(DA_WQ, 512, 0, 512, 512);
  const _Float16* w16_da_k   = cvt(DA_WK, 512, 0, 512, 512);
  const _Float16* w16_da_v   = cvt(DA_WV, 512, 0, 512, 512);
  const _Float16* w16_da_o   = cvt(DA_WO, 512, 0, 512, 512);
  const _Float16* w16_da_1   = cvt(DA_W1, 512, 0, 2048, 512);
  const _Float16* w16_da_2   = cvt(DA_W2, 2048, 0, 512, 2048);
  const _Float16* w16_inw[4];
  const _Float16* w16_xp[4];
  const _Float16* w16_dt[4];
  const _Float16* w16_ow[4];
  for (int i = 0; i < 4; ++i) {
    w16_inw[i] = wcur;
    {
      long total = (long)2048 * 512;
      k_cvt_w<<<dim3(2048), dim3(256), 0, stream>>>(
          FIN(M_INW) + (size_t)i * 2048 * 512, 512, 0, wcur, total, 512);
      wcur += total;
    }
    w16_xp[i] = wcur;
    {
      long total = (long)64 * 1024;
      k_cvt_w<<<dim3(256), dim3(256), 0, stream>>>(
          FIN(M_XPROJW) + (size_t)i * 64 * 1024, 1024, 0, wcur, total, 1024);
      wcur += total;
    }
    w16_dt[i] = wcur;
    {
      long total = (long)1024 * 32;
      k_cvt_w<<<dim3(128), dim3(256), 0, stream>>>(
          FIN(M_DTW) + (size_t)i * 1024 * 32, 32, 0, wcur, total, 32);
      wcur += total;
    }
    w16_ow[i] = wcur;
    {
      long total = (long)512 * 1024;
      k_cvt_w<<<dim3(2048), dim3(256), 0, stream>>>(
          FIN(M_OUTW) + (size_t)i * 512 * 1024, 1024, 0, wcur, total, 1024);
      wcur += total;
    }
  }

  // ---- encoder: enc_kv = concat(x, pf, xyz) @ enc_ad_w.T + b ----
  gemm(stream, FIN(IN_PF), 512, 0, w16_enc_ad, FIN(P_ENC_AD_B), nullptr, 0,
       S0, 512, MBIG, 512, 512, 0);
  k_encode_fixup<<<dim3(MBIG), dim3(256), 0, stream>>>(
      FIN(IN_X), FIN(IN_XYZ), FIN(P_ENC_AD_W), S0);

  // ---- enc attention K/V from LN(enc_kv) ----
  lnorm(stream, S0, FIN(EA_KVLNG), FIN(EA_KVLNB), S1, MBIG);
  gemm(stream, S1, 512, 0, w16_ea_k, FIN(EA_BK), nullptr, 0,
       S2, 512, MBIG, 512, 512, 0);
  gemm(stream, S1, 512, 0, w16_ea_v, FIN(EA_BV), nullptr, 0,
       S3, 512, MBIG, 512, 512, 0);

  // ---- enc_q = [lang_proj ; time_proj] per batch ----
  for (int b = 0; b < Bb; ++b)
    gemm(stream, FIN(IN_LANG) + (size_t)b * LTXT * 512, 512, 0,
         w16_lang, FIN(P_LANG_B), nullptr, 0,
         cur + (size_t)b * LQ * 512, 512, LTXT, 512, 512, 0);
  gemm(stream, FIN(IN_TEMB), 512, 0, w16_time, FIN(P_TIME_B), nullptr, 0,
       cur + (size_t)LTXT * 512, LQ * 512, Bb, 512, 512, 0);

  // ---- enc cross-attention (78 queries x 8192 keys) ----
  lnorm(stream, cur, FIN(EA_QLNG), FIN(EA_QLNB), tmpa, MSMALL);
  gemm(stream, tmpa, 512, 0, w16_ea_q, FIN(EA_BQ), nullptr, 0,
       tmpb, 512, MSMALL, 512, 512, 0);
  k_attn_qsmall<<<dim3(Bb * HEADS * LQ), dim3(256), 0, stream>>>(
      tmpb, S2, S3, tmpc, LQ, Nn);
  gemm(stream, tmpc, 512, 0, w16_ea_o, FIN(EA_BO), cur, 512,
       tmph, 512, MSMALL, 512, 512, 0);
  lnorm(stream, tmph, FIN(EA_MLNG), FIN(EA_MLNB), tmpm, MSMALL);
  gemm(stream, tmpm, 512, 0, w16_ea_1, FIN(EA_B1), nullptr, 0,
       tmpw, 2048, MSMALL, 2048, 512, 1);
  gemm(stream, tmpw, 2048, 0, w16_ea_2, FIN(EA_B2), tmph, 512,
       cur, 512, MSMALL, 512, 2048, 0);

  // ---- 4 mamba layers on [B,78,512] ----
  float* mcur = cur;
  float* mnext = curnew;
  for (int i = 0; i < 4; ++i) {
    gemm(stream, mcur, 512, 0, w16_inw[i], nullptr, nullptr, 0,
         xzbuf, 2048, MSMALL, 2048, 512, 0);
    k_mamba_conv<<<dim3(Bb * 4), dim3(256), 0, stream>>>(
        xzbuf, FIN(M_CONVW) + (size_t)i * DINNER * 4,
        FIN(M_CONVB) + (size_t)i * DINNER, xcbuf);
    gemm(stream, xcbuf, 1024, 0, w16_xp[i], nullptr, nullptr, 0,
         xdblbf, 64, MSMALL, 64, 1024, 0);
    gemm(stream, xdblbf, 64, 0, w16_dt[i], FIN(M_DTB) + (size_t)i * DINNER,
         nullptr, 0, dtbuf, 1024, MSMALL, 1024, DTRANK, 3);
    k_mamba_scan<<<dim3(Bb * 4), dim3(256), 0, stream>>>(
        xcbuf, xdblbf, dtbuf, xzbuf,
        FIN(M_ALOG) + (size_t)i * DINNER * DSTATE,
        FIN(M_DD) + (size_t)i * DINNER, ybuf);
    gemm(stream, ybuf, 1024, 0, w16_ow[i], nullptr, nullptr, 0,
         mnext, 512, MSMALL, 512, 1024, 0);
    float* t = mcur; mcur = mnext; mnext = t;
  }
  // after even number of swaps, mcur holds the mamba output

  // ---- decoder: dec_q = enc_kv @ dec_ad_w.T + b ----
  gemm(stream, S0, 512, 0, w16_dec_ad, FIN(P_DEC_AD_B), nullptr, 0,
       S1, 512, MBIG, 512, 512, 0);

  // dec attention K/V from LN(mamba out) — tiny
  lnorm(stream, mcur, FIN(DA_KVLNG), FIN(DA_KVLNB), tmpa, MSMALL);
  gemm(stream, tmpa, 512, 0, w16_da_k, FIN(DA_BK), nullptr, 0,
       kv2k, 512, MSMALL, 512, 512, 0);
  gemm(stream, tmpa, 512, 0, w16_da_v, FIN(DA_BV), nullptr, 0,
       kv2v, 512, MSMALL, 512, 512, 0);

  // dec Q (32768 rows)
  lnorm(stream, S1, FIN(DA_QLNG), FIN(DA_QLNB), S2, MBIG);
  gemm(stream, S2, 512, 0, w16_da_q, FIN(DA_BQ), nullptr, 0,
       S3, 512, MBIG, 512, 512, 0);

  // dec cross-attention (8192 queries x 78 keys)
  k_attn_ksmall<<<dim3(Bb * HEADS * (Nn / 128)), dim3(128), 0, stream>>>(
      S3, kv2k, kv2v, S2, Nn, LQ);

  // h2 = dec_q + o @ wo.T + bo   (S0: enc_kv now dead)
  gemm(stream, S2, 512, 0, w16_da_o, FIN(DA_BO), S1, 512,
       S0, 512, MBIG, 512, 512, 0);

  // out = h2 + GELU(LN(h2) @ w1.T + b1) @ w2.T + b2, M chunked 4x8192
  lnorm(stream, S0, FIN(DA_MLNG), FIN(DA_MLNB), S3, MBIG);
  for (int c = 0; c < 4; ++c) {
    size_t M0 = (size_t)c * 8192;
    gemm(stream, S3 + M0 * 512, 512, 0, w16_da_1, FIN(DA_B1),
         nullptr, 0, S4, 2048, 8192, 2048, 512, 1);
    gemm(stream, S4, 2048, 0, w16_da_2, FIN(DA_B2),
         S0 + M0 * 512, 512, dout + M0 * 512, 512, 8192, 512, 2048, 0);
  }
#undef FIN
}
